// LlamaLikeAttention_22883585753585
// MI455X (gfx1250) — matrix operand order repaired and checked by reading, not compiled, not run on hardware
//
#include <hip/hip_runtime.h>
#include <hip/hip_bf16.h>
#include <math.h>

#define S_LEN 2048
#define DMODEL 2048
#define NHEADS 16
#define HDIM 128

typedef __bf16 bf16;
typedef __attribute__((ext_vector_type(16))) __bf16 v16bf;
typedef __attribute__((ext_vector_type(8)))  float  v8f;
typedef __attribute__((vector_size(16))) int v4i;   // matches builtin proto

union AB16 { v16bf v; bf16 h[16]; uint4 q[2]; };
union ACC  { v8f  v; float f[8]; };

__device__ __forceinline__ v8f wmma_bf16(v16bf a, v16bf b, v8f c) {
  // D = A(16x32 bf16) * B(32x16 bf16) + C(16x16 f32)
  return __builtin_amdgcn_wmma_f32_16x16x32_bf16(
      /*neg_a=*/false, a, /*neg_b=*/false, b,
      /*c_mod=*/(short)0, c, /*reuse_a=*/false, /*reuse_b=*/false);
}

// ---- CDNA5 async global->LDS copy (ASYNCcnt-tracked), with sync fallback ----
#if __has_builtin(__builtin_amdgcn_global_load_async_to_lds_b128) && \
    __has_builtin(__builtin_amdgcn_s_wait_asynccnt)
#define HAVE_ASYNC_LDS 1
#define WAIT_ASYNC(n) __builtin_amdgcn_s_wait_asynccnt(n)
#else
#define HAVE_ASYNC_LDS 0
#define WAIT_ASYNC(n)
#endif

__device__ __forceinline__ void async_cp16(const bf16* g, bf16* l) {
#if HAVE_ASYNC_LDS
  __builtin_amdgcn_global_load_async_to_lds_b128(
      (__attribute__((address_space(1))) v4i*)g,
      (__attribute__((address_space(3))) v4i*)l, 0, 0);
#else
  *(uint4*)l = *(const uint4*)g;
#endif
}

// ---------------------------------------------------------------------------
// fp32 -> bf16 cast
// ---------------------------------------------------------------------------
__global__ __launch_bounds__(256) void cast_f32_bf16(
    const float* __restrict__ src, bf16* __restrict__ dst, int n) {
  int i = blockIdx.x * 256 + threadIdx.x;
  if (i < n) dst[i] = (bf16)src[i];
}

// ---------------------------------------------------------------------------
// fp32 (RxC) -> bf16 transposed (CxR): dst[c][r] = (bf16)src[r][c]
// ---------------------------------------------------------------------------
__global__ __launch_bounds__(256) void cast_transpose_f32_bf16(
    const float* __restrict__ src, bf16* __restrict__ dst, int R, int Cc) {
  __shared__ float tile[32][33];
  const int bx = blockIdx.x * 32;   // column base
  const int by = blockIdx.y * 32;   // row base
  const int tx = threadIdx.x & 31;
  const int ty = threadIdx.x >> 5;  // 0..7
#pragma unroll
  for (int i = 0; i < 32; i += 8)
    tile[ty + i][tx] = src[(size_t)(by + ty + i) * Cc + bx + tx];
  __syncthreads();
#pragma unroll
  for (int i = 0; i < 32; i += 8)
    dst[(size_t)(bx + ty + i) * R + by + tx] = (bf16)tile[tx][ty + i];
}

// ---------------------------------------------------------------------------
// C(MxN,f32) = A(MxK,bf16 rowmajor) @ Bt(NxK,bf16 rowmajor)^T
// 256 threads (8 waves), block tile 128x128, K-step 32, double-buffered
// async global->LDS fills. Wave (wm,wn) computes a 32x64 sub-tile: 8 WMMA.
// ---------------------------------------------------------------------------
__global__ __launch_bounds__(256) void gemm_bf16f32_tn(
    const bf16* __restrict__ A, const bf16* __restrict__ Bt,
    float* __restrict__ C, int M, int N, int K) {
  __shared__ __align__(16) bf16 As[2][128][32];   // [buf][m][k]   8KB/buf
  __shared__ __align__(16) bf16 Bs[2][128][32];   // [buf][n][k]   8KB/buf

  const int n0 = blockIdx.x * 128;
  const int m0 = blockIdx.y * 128;
  const int tid = threadIdx.x;
  const int wave = tid >> 5;
  const int lane = tid & 31;
  const int mrow = lane & 15;
  const int hh   = lane >> 4;          // lane half
  const int wm = (wave & 3) * 32;      // wave M offset in tile
  const int wn = (wave >> 2) * 64;     // wave N offset in tile

  // per-thread fill slots: row tid>>1, 16 halfs starting at (tid&1)*16
  const int fr = tid >> 1;
  const int fc = (tid & 1) * 16;

  ACC acc[2][4];
#pragma unroll
  for (int mi = 0; mi < 2; ++mi)
#pragma unroll
    for (int nt = 0; nt < 4; ++nt)
#pragma unroll
      for (int r = 0; r < 8; ++r) acc[mi][nt].f[r] = 0.f;

  auto fill = [&](int buf, int k0) {
    const bf16* ga = A  + (size_t)(m0 + fr) * K + k0 + fc;
    const bf16* gb = Bt + (size_t)(n0 + fr) * K + k0 + fc;
    async_cp16(ga,     &As[buf][fr][fc]);
    async_cp16(ga + 8, &As[buf][fr][fc + 8]);
    async_cp16(gb,     &Bs[buf][fr][fc]);
    async_cp16(gb + 8, &Bs[buf][fr][fc + 8]);
  };

  const int nsteps = K >> 5;
  fill(0, 0);
  for (int i = 0; i < nsteps; ++i) {
    const int buf = i & 1;
    if (i + 1 < nsteps) {
      fill(buf ^ 1, (i + 1) * 32);
      WAIT_ASYNC(4);          // current stage's 4 copies complete (in-order)
    } else {
      WAIT_ASYNC(0);
    }
    __syncthreads();

    AB16 a[2];
    const int kbA = hh * 8;
    a[0].q[0] = *(const uint4*)(&As[buf][wm + mrow][kbA]);
    a[0].q[1] = *(const uint4*)(&As[buf][wm + mrow][16 + kbA]);
    a[1].q[0] = *(const uint4*)(&As[buf][wm + 16 + mrow][kbA]);
    a[1].q[1] = *(const uint4*)(&As[buf][wm + 16 + mrow][16 + kbA]);

    const int kbB = hh * 8;
#pragma unroll
    for (int nt = 0; nt < 4; ++nt) {
      AB16 b;
      const int bn = wn + nt * 16 + mrow;
      b.q[0] = *(const uint4*)(&Bs[buf][bn][kbB]);
      b.q[1] = *(const uint4*)(&Bs[buf][bn][kbB + 16]);
      acc[0][nt].v = wmma_bf16(a[0].v, b.v, acc[0][nt].v);
      acc[1][nt].v = wmma_bf16(a[1].v, b.v, acc[1][nt].v);
    }
    __syncthreads();
  }

#pragma unroll
  for (int mi = 0; mi < 2; ++mi)
#pragma unroll
    for (int nt = 0; nt < 4; ++nt)
#pragma unroll
      for (int r = 0; r < 8; ++r)
        C[(size_t)(m0 + wm + mi * 16 + r + 8 * hh) * N + n0 + wn + nt * 16 +
          mrow] = acc[mi][nt].f[r];
}

// ---------------------------------------------------------------------------
// RoPE on Qf/Kf (fp32) -> Qb/Kb (bf16, [s][h*128+d]); Vf -> Vt (bf16, [h*128+d][s])
// ---------------------------------------------------------------------------
__global__ __launch_bounds__(256) void rope_cast(
    const float* __restrict__ Qf, const float* __restrict__ Kf,
    const float* __restrict__ Vf, const float* __restrict__ fc,
    const float* __restrict__ fs, bf16* __restrict__ Qb,
    bf16* __restrict__ Kb, bf16* __restrict__ Vt) {
  int idx = blockIdx.x * 256 + threadIdx.x;
  int i = idx & 63;
  int h = (idx >> 6) & (NHEADS - 1);
  int s = idx >> 10;
  if (s >= S_LEN) return;
  float c = fc[s * 64 + i];
  float sn = fs[s * 64 + i];
  size_t base = (size_t)s * DMODEL + h * HDIM + 2 * i;
  float q0 = Qf[base], q1 = Qf[base + 1];
  Qb[base]     = (bf16)(q0 * c - q1 * sn);
  Qb[base + 1] = (bf16)(q0 * sn + q1 * c);
  float k0 = Kf[base], k1 = Kf[base + 1];
  Kb[base]     = (bf16)(k0 * c - k1 * sn);
  Kb[base + 1] = (bf16)(k0 * sn + k1 * c);
  float v0 = Vf[base], v1 = Vf[base + 1];
  Vt[(size_t)(h * HDIM + 2 * i) * S_LEN + s]     = (bf16)v0;
  Vt[(size_t)(h * HDIM + 2 * i + 1) * S_LEN + s] = (bf16)v1;
}

// ---------------------------------------------------------------------------
// Flash attention with softcap-tanh + causal mask. One wave (32 threads) per
// (16-query-row block, head). All matmuls via v_wmma_f32_16x16x32_bf16.
// ---------------------------------------------------------------------------
__global__ __launch_bounds__(32) void attn_flash(
    const bf16* __restrict__ Qb, const bf16* __restrict__ Kb,
    const bf16* __restrict__ Vt, float* __restrict__ O) {
  __shared__ __align__(16) float Ps[16][32];   // P staging: C-layout -> A-layout

  const int q0 = blockIdx.x * 16;
  const int h = blockIdx.y;
  const int lane = threadIdx.x;
  const int mrow = lane & 15;
  const int hh   = lane >> 4;
  const float scale = 0.08838834764831845f;    // 1/sqrt(128)

  AB16 qf[4];
  {
    const bf16* qrow = Qb + (size_t)(q0 + mrow) * DMODEL + h * HDIM;
    const int kbA = hh * 8;
#pragma unroll
    for (int c = 0; c < 4; ++c) {
      qf[c].q[0] = *(const uint4*)(qrow + c * 32 + kbA);
      qf[c].q[1] = *(const uint4*)(qrow + c * 32 + 16 + kbA);
    }
  }

  float m_i[8], l_i[8];
  ACC o[8];
  for (int r = 0; r < 8; ++r) { m_i[r] = -INFINITY; l_i[r] = 0.f; }
  for (int d = 0; d < 8; ++d)
    for (int r = 0; r < 8; ++r) o[d].f[r] = 0.f;

  const int jmax = (q0 + 15) >> 5;
  const int kb2 = hh * 8;

  for (int j = 0; j <= jmax; ++j) {
    const int kbeg = j * 32;
    if (j < jmax) {
      __builtin_prefetch(Kb + (size_t)(kbeg + 32 + mrow) * DMODEL + h * HDIM, 0, 3);
      __builtin_prefetch(Vt + (size_t)(h * HDIM + mrow) * S_LEN + kbeg + 32, 0, 3);
    }

    // scores: S(16x32) = Q(16x128) @ K^T as two 16x16 C-tiles
    ACC st[2];
#pragma unroll
    for (int t = 0; t < 2; ++t) {
      for (int r = 0; r < 8; ++r) st[t].f[r] = 0.f;
      const bf16* krow = Kb + (size_t)(kbeg + t * 16 + mrow) * DMODEL + h * HDIM;
#pragma unroll
      for (int c = 0; c < 4; ++c) {
        AB16 kf;
        kf.q[0] = *(const uint4*)(krow + c * 32 + kb2);
        kf.q[1] = *(const uint4*)(krow + c * 32 + kb2 + 16);
        st[t].v = wmma_bf16(qf[c].v, kf.v, st[t].v);
      }
    }

    // softcap + causal mask + online softmax
#pragma unroll
    for (int r = 0; r < 8; ++r) {
      const int mq = q0 + r + 8 * hh;
      float s0 = st[0].f[r] * scale; s0 = 50.f * tanhf(s0 * 0.02f);
      float s1 = st[1].f[r] * scale; s1 = 50.f * tanhf(s1 * 0.02f);
      if (kbeg + mrow > mq)      s0 = -INFINITY;
      if (kbeg + 16 + mrow > mq) s1 = -INFINITY;
      float rowm = fmaxf(s0, s1);
#pragma unroll
      for (int msk = 1; msk < 16; msk <<= 1)
        rowm = fmaxf(rowm, __shfl_xor(rowm, msk, 32));
      float mnew = fmaxf(m_i[r], rowm);
      float corr = __expf(m_i[r] - mnew);
      float p0 = __expf(s0 - mnew);
      float p1 = __expf(s1 - mnew);
      float rs = p0 + p1;
#pragma unroll
      for (int msk = 1; msk < 16; msk <<= 1) rs += __shfl_xor(rs, msk, 32);
      l_i[r] = l_i[r] * corr + rs;
      m_i[r] = mnew;
#pragma unroll
      for (int d = 0; d < 8; ++d) o[d].f[r] *= corr;
      Ps[r + 8 * hh][mrow]      = p0;
      Ps[r + 8 * hh][16 + mrow] = p1;
    }
    __syncthreads();   // single-wave WG: s_nop + compiler fence

    // P (C-layout f32) -> A-layout bf16 via LDS
    AB16 pf;
    const int kbA = hh * 8;
#pragma unroll
    for (int e = 0; e < 8; ++e) {
      pf.h[e]     = (bf16)Ps[mrow][kbA + e];
      pf.h[8 + e] = (bf16)Ps[mrow][16 + kbA + e];
    }

    // O += P(16x32) @ V(32x128)
#pragma unroll
    for (int d = 0; d < 8; ++d) {
      AB16 vf;
      const bf16* vrow = Vt + (size_t)(h * HDIM + d * 16 + mrow) * S_LEN + kbeg;
      vf.q[0] = *(const uint4*)(vrow + kb2);
      vf.q[1] = *(const uint4*)(vrow + kb2 + 16);
      o[d].v = wmma_bf16(pf.v, vf.v, o[d].v);
    }
    __syncthreads();
  }

#pragma unroll
  for (int d = 0; d < 8; ++d) {
#pragma unroll
    for (int r = 0; r < 8; ++r) {
      float inv = 1.f / l_i[r];
      O[(size_t)(q0 + r + 8 * hh) * DMODEL + h * HDIM + d * 16 + mrow] =
          o[d].f[r] * inv;
    }
  }
}

// ---------------------------------------------------------------------------
extern "C" void kernel_launch(void* const* d_in, const int* in_sizes, int n_in,
                              void* d_out, int out_size, void* d_ws,
                              size_t ws_size, hipStream_t stream) {
  const float* x  = (const float*)d_in[0];
  const float* wq = (const float*)d_in[1];
  const float* wk = (const float*)d_in[2];
  const float* wv = (const float*)d_in[3];
  const float* wo = (const float*)d_in[4];
  const float* fc = (const float*)d_in[5];
  const float* fs = (const float*)d_in[6];
  float* out = (float*)d_out;

  const size_t E  = (size_t)S_LEN * DMODEL;
  const size_t MB = 1024 * 1024;
  char* ws = (char*)d_ws;
  bf16* xb   = (bf16*)(ws + 0 * MB);
  bf16* wqbT = (bf16*)(ws + 8 * MB);    // wq^T, bf16, N x K
  bf16* wkbT = (bf16*)(ws + 16 * MB);
  bf16* wvbT = (bf16*)(ws + 24 * MB);
  bf16* wobT = (bf16*)(ws + 32 * MB);
  float* Qf = (float*)(ws + 40 * MB);
  float* Kf = (float*)(ws + 56 * MB);
  float* Vf = (float*)(ws + 72 * MB);
  bf16* Qb  = (bf16*)(ws + 88 * MB);
  bf16* Kb  = (bf16*)(ws + 96 * MB);
  bf16* Vt  = (bf16*)(ws + 104 * MB);
  float* O  = (float*)(ws + 40 * MB);   // reuse Qf (consumed by rope_cast)
  bf16* Ob  = (bf16*)(ws + 56 * MB);    // reuse Kf (consumed by rope_cast)

  const int nE = (int)E;
  const int cg = (nE + 255) / 256;
  cast_f32_bf16<<<cg, 256, 0, stream>>>(x, xb, nE);
  dim3 tg(DMODEL / 32, DMODEL / 32);
  cast_transpose_f32_bf16<<<tg, 256, 0, stream>>>(wq, wqbT, DMODEL, DMODEL);
  cast_transpose_f32_bf16<<<tg, 256, 0, stream>>>(wk, wkbT, DMODEL, DMODEL);
  cast_transpose_f32_bf16<<<tg, 256, 0, stream>>>(wv, wvbT, DMODEL, DMODEL);
  cast_transpose_f32_bf16<<<tg, 256, 0, stream>>>(wo, wobT, DMODEL, DMODEL);

  dim3 gg(DMODEL / 128, S_LEN / 128);
  gemm_bf16f32_tn<<<gg, 256, 0, stream>>>(xb, wqbT, Qf, S_LEN, DMODEL, DMODEL);
  gemm_bf16f32_tn<<<gg, 256, 0, stream>>>(xb, wkbT, Kf, S_LEN, DMODEL, DMODEL);
  gemm_bf16f32_tn<<<gg, 256, 0, stream>>>(xb, wvbT, Vf, S_LEN, DMODEL, DMODEL);

  rope_cast<<<(S_LEN * NHEADS * 64) / 256, 256, 0, stream>>>(
      Qf, Kf, Vf, fc, fs, Qb, Kb, Vt);

  attn_flash<<<dim3(S_LEN / 16, NHEADS), 32, 0, stream>>>(Qb, Kb, Vt, O);

  cast_f32_bf16<<<cg, 256, 0, stream>>>(O, Ob, nE);
  gemm_bf16f32_tn<<<gg, 256, 0, stream>>>(Ob, wobT, out, S_LEN, DMODEL, DMODEL);
}
